// ScaledDotProductAttention_19550691132117
// MI455X (gfx1250) — matrix operand order repaired and checked by reading, not compiled, not run on hardware
//
#include <hip/hip_runtime.h>

typedef __attribute__((ext_vector_type(16))) _Float16 v16h;
typedef __attribute__((ext_vector_type(8)))  _Float16 v8h;
typedef __attribute__((ext_vector_type(2)))  _Float16 v2h;
typedef __attribute__((ext_vector_type(8)))  float    v8f;

#define B_   2
#define H_   16
#define S_   2048
#define DK_  64
// Fold softmax scale (1/sqrt(64)) AND log2(e) into Q once:
// exp(s/8) == exp2( (q*QSCALE_) . k )
#define QSCALE_  0.18033688011112042f   // 0.125 * log2(e)
#define NEG_BIG_ (-1.0e9f)              // exp2(-1e9) == 0
#define VSTRIDE_ 48                     // f16 units; 96B rows keep 32B alignment

__device__ __forceinline__ v8f wmma16(v16h a, v16h b, v8f c) {
  // D = A(16x32 f16) * B(32x16 f16) + C(16x16 f32)
  return __builtin_amdgcn_wmma_f32_16x16x32_f16(
      /*neg_a=*/false, a, /*neg_b=*/false, b,
      /*c_mod=*/(short)0, c, /*reuse_a=*/false, /*reuse_b=*/false);
}

// A-operand fragment (16x32 f16) from fp32 row-major global memory, pre-scaled.
// A layout: K = 16*half + 2v (contiguous).
__device__ __forceinline__ v16h load_a_scaled(const float* __restrict__ base,
                                              int row, int kbase, int half,
                                              float scale) {
  v16h a;
  const float* rp = base + (size_t)row * DK_;
#pragma unroll
  for (int v = 0; v < 8; ++v) {
    int k0 = kbase + 16 * half + 2 * v;
    a[2 * v]     = (_Float16)(rp[k0] * scale);
    a[2 * v + 1] = (_Float16)(rp[k0 + 1] * scale);
  }
  return a;
}

// mask + exp2 + packed f16 store of one 16x32 P tile (interleaved k' layout)
__device__ __forceinline__ void softmax_store(v8f s0, v8f s1,
                                              const int* __restrict__ Mb,
                                              int rowbase, int kb,
                                              int ln, int half,
                                              _Float16* pw) {
#pragma unroll
  for (int r = 0; r < 8; ++r) {
    const int row = rowbase + r + 8 * half;
    const size_t mrow = (size_t)row * S_;
    float x0 = (Mb[mrow + kb + ln]      != 0) ? s0[r] : NEG_BIG_;
    float x1 = (Mb[mrow + kb + 16 + ln] != 0) ? s1[r] : NEG_BIG_;
    float p0 = __builtin_amdgcn_exp2f(x0);   // == exp(score/8)
    float p1 = __builtin_amdgcn_exp2f(x1);
    v2h pp;
    pp[0] = (_Float16)p0;                    // k' = 2*ln
    pp[1] = (_Float16)p1;                    // k' = 2*ln+1
    *(v2h*)&pw[(r + 8 * half) * 32 + 2 * ln] = pp;
  }
}

// rebuild a P tile as an A-operand (two contiguous 16B runs per lane)
__device__ __forceinline__ v16h rebuild_pa(const _Float16* pw, int ln, int half) {
  v16h pa;
  v8h lo = *(const v8h*)&pw[ln * 32 + 16 * half];
  v8h hi = *(const v8h*)&pw[ln * 32 + 16 * half + 8];
#pragma unroll
  for (int i = 0; i < 8; ++i) { pa[i] = lo[i]; pa[8 + i] = hi[i]; }
  return pa;
}

__global__ __launch_bounds__(128)
void fa_fwd_kernel(const float* __restrict__ Q, const float* __restrict__ K,
                   const float* __restrict__ V, const int* __restrict__ M,
                   float* __restrict__ O) {
  const int qt   = blockIdx.x;           // 0..15  (128-row Q tile)
  const int h    = blockIdx.y;
  const int b    = blockIdx.z;
  const int tid  = threadIdx.x;
  const int wave = tid >> 5;             // 0..3
  const int lane = tid & 31;
  const int half = lane >> 4;            // 0/1
  const int ln   = lane & 15;

  const size_t head_off = ((size_t)(b * H_ + h)) * S_ * DK_;
  const float* Qh = Q + head_off;
  const float* Kh = K + head_off;
  const float* Vh = V + head_off;
  float*       Oh = O + head_off;
  const int*   Mb = M + (size_t)b * S_ * S_;

  const int q0 = qt * 128 + wave * 32;   // wave owns 32 query rows (two 16-row sets)

  // double-buffered cooperative staging (f16):
  //   Ksh : K tile row-major [key][dk]
  //   Vsh : V tile transposed [dk][k'] with INTERLEAVED key order
  //         k' = 2c -> physical key c (s0 cols); k' = 2c+1 -> key c+16 (s1 cols)
  __shared__ __attribute__((aligned(64))) _Float16 Ksh[2][32][64];        // 8 KB
  __shared__ __attribute__((aligned(64))) _Float16 Vsh[2][64][VSTRIDE_];  // 12 KB
  __shared__ __attribute__((aligned(64))) _Float16 psh[4][2][16 * 32];    // 8 KB

  // staging index maps
  const int skey = tid >> 2;            // 0..31 : key row for K staging
  const int sdkb = (tid & 3) << 4;      // 0/16/32/48 : dk base for K staging
  const int vc   = tid & 15;            // key pair (c, c+16) for V staging
  const int vdkb = (tid >> 4) * 8;      // dk base (8-wide) for V staging

  // Q fragments: two row sets, pre-scaled by 0.125*log2(e), loaded once.
  v16h aqA0 = load_a_scaled(Qh, q0 + ln,      0,  half, QSCALE_);
  v16h aqA1 = load_a_scaled(Qh, q0 + ln,      32, half, QSCALE_);
  v16h aqB0 = load_a_scaled(Qh, q0 + 16 + ln, 0,  half, QSCALE_);
  v16h aqB1 = load_a_scaled(Qh, q0 + 16 + ln, 32, half, QSCALE_);

  // constant all-ones B-fragment: o4 = P * ones == row-sum of P (softmax denom)
  v16h vone;
#pragma unroll
  for (int i = 0; i < 16; ++i) vone[i] = (_Float16)1.0f;

  v8f oA0 = {}, oA1 = {}, oA2 = {}, oA3 = {}, oA4 = {};
  v8f oB0 = {}, oB1 = {}, oB2 = {}, oB3 = {}, oB4 = {};

  for (int kb = 0; kb < S_; kb += 32) {
    const int it = (kb >> 5) & 1;

    if (kb + 32 < S_) {  // uniform: prefetch next block
      __builtin_prefetch(Kh + (size_t)(kb + 32 + skey) * DK_ + sdkb, 0, 3);
      __builtin_prefetch(Vh + (size_t)(kb + 32 + skey) * DK_ + sdkb, 0, 3);
    }

    // ---- cooperative stage K tile (32x64) as f16, row-major ----
    {
      const float* kr = Kh + (size_t)(kb + skey) * DK_ + sdkb;
      float4 k0 = ((const float4*)kr)[0];
      float4 k1 = ((const float4*)kr)[1];
      float4 k2 = ((const float4*)kr)[2];
      float4 k3 = ((const float4*)kr)[3];
      v8h h0, h1;
      h0[0] = (_Float16)k0.x; h0[1] = (_Float16)k0.y;
      h0[2] = (_Float16)k0.z; h0[3] = (_Float16)k0.w;
      h0[4] = (_Float16)k1.x; h0[5] = (_Float16)k1.y;
      h0[6] = (_Float16)k1.z; h0[7] = (_Float16)k1.w;
      h1[0] = (_Float16)k2.x; h1[1] = (_Float16)k2.y;
      h1[2] = (_Float16)k2.z; h1[3] = (_Float16)k2.w;
      h1[4] = (_Float16)k3.x; h1[5] = (_Float16)k3.y;
      h1[6] = (_Float16)k3.z; h1[7] = (_Float16)k3.w;
      *(v8h*)&Ksh[it][skey][sdkb]     = h0;
      *(v8h*)&Ksh[it][skey][sdkb + 8] = h1;
    }
    // ---- cooperative stage V tile (32x64) as f16, transposed + key-interleaved ----
    {
      const float* vr0 = Vh + (size_t)(kb + vc) * DK_ + vdkb;        // key c
      const float* vr1 = vr0 + 16 * DK_;                              // key c+16
      float4 a0 = ((const float4*)vr0)[0];
      float4 a1 = ((const float4*)vr0)[1];
      float4 b0 = ((const float4*)vr1)[0];
      float4 b1 = ((const float4*)vr1)[1];
      float va[8] = {a0.x, a0.y, a0.z, a0.w, a1.x, a1.y, a1.z, a1.w};
      float vb[8] = {b0.x, b0.y, b0.z, b0.w, b1.x, b1.y, b1.z, b1.w};
#pragma unroll
      for (int i = 0; i < 8; ++i) {
        v2h pr;
        pr[0] = (_Float16)va[i];   // k' = 2c
        pr[1] = (_Float16)vb[i];   // k' = 2c+1
        *(v2h*)&Vsh[it][vdkb + i][2 * vc] = pr;   // one b32 store
      }
    }
    __syncthreads();

    // ---- S' = (Q*QSCALE)*K^T : K fragments loaded once, reused by both row sets
    v16h kb00 = *(const v16h*)&Ksh[it][ln][16 * half];
    v16h kb01 = *(const v16h*)&Ksh[it][ln][32 + 16 * half];
    v16h kb10 = *(const v16h*)&Ksh[it][16 + ln][16 * half];
    v16h kb11 = *(const v16h*)&Ksh[it][16 + ln][32 + 16 * half];

    v8f z = {};
    v8f sA0 = wmma16(aqA0, kb00, z);  sA0 = wmma16(aqA1, kb01, sA0);
    v8f sA1 = wmma16(aqA0, kb10, z);  sA1 = wmma16(aqA1, kb11, sA1);
    v8f sB0 = wmma16(aqB0, kb00, z);  sB0 = wmma16(aqB1, kb01, sB0);
    v8f sB1 = wmma16(aqB0, kb10, z);  sB1 = wmma16(aqB1, kb11, sB1);

    // ---- mask + exp2, packed f16 P stores for both row sets ----
    softmax_store(sA0, sA1, Mb, q0,      kb, ln, half, &psh[wave][0][0]);
    softmax_store(sB0, sB1, Mb, q0 + 16, kb, ln, half, &psh[wave][1][0]);

    // cross-lane DS dependency: all P stores must land before A-layout reads
    asm volatile("s_wait_dscnt 0" ::: "memory");

    v16h paA = rebuild_pa(&psh[wave][0][0], ln, half);
    v16h paB = rebuild_pa(&psh[wave][1][0], ln, half);

    // ---- O += P*V ; denominators on the matrix pipe; V fragments reused 2x ----
    v16h vb0 = *(const v16h*)&Vsh[it][0  + ln][16 * half];
    v16h vb1 = *(const v16h*)&Vsh[it][16 + ln][16 * half];
    v16h vb2 = *(const v16h*)&Vsh[it][32 + ln][16 * half];
    v16h vb3 = *(const v16h*)&Vsh[it][48 + ln][16 * half];

    oA0 = wmma16(paA, vb0, oA0);
    oB0 = wmma16(paB, vb0, oB0);
    oA1 = wmma16(paA, vb1, oA1);
    oB1 = wmma16(paB, vb1, oB1);
    oA2 = wmma16(paA, vb2, oA2);
    oB2 = wmma16(paB, vb2, oB2);
    oA3 = wmma16(paA, vb3, oA3);
    oB3 = wmma16(paB, vb3, oB3);
    oA4 = wmma16(paA, vone, oA4);   // row-sum of P, set A
    oB4 = wmma16(paB, vone, oB4);   // row-sum of P, set B
  }

  // ---- final: l already reduced by the ones-WMMAs; normalize + store fp32 ----
#pragma unroll
  for (int r = 0; r < 8; ++r) {
    {
      const float invl = 1.0f / oA4[r];
      const int row = q0 + r + 8 * half;
      float* orow = Oh + (size_t)row * DK_;
      orow[ln]      = oA0[r] * invl;
      orow[16 + ln] = oA1[r] * invl;
      orow[32 + ln] = oA2[r] * invl;
      orow[48 + ln] = oA3[r] * invl;
    }
    {
      const float invl = 1.0f / oB4[r];
      const int row = q0 + 16 + r + 8 * half;
      float* orow = Oh + (size_t)row * DK_;
      orow[ln]      = oB0[r] * invl;
      orow[16 + ln] = oB1[r] * invl;
      orow[32 + ln] = oB2[r] * invl;
      orow[48 + ln] = oB3[r] * invl;
    }
  }
}

extern "C" void kernel_launch(void* const* d_in, const int* in_sizes, int n_in,
                              void* d_out, int out_size, void* d_ws, size_t ws_size,
                              hipStream_t stream) {
  (void)in_sizes; (void)n_in; (void)out_size; (void)d_ws; (void)ws_size;
  const float* Q = (const float*)d_in[0];
  const float* K = (const float*)d_in[1];
  const float* V = (const float*)d_in[2];
  const int*   M = (const int*)d_in[3];
  float*       O = (float*)d_out;

  dim3 grid(S_ / 128, H_, B_);  // (16, 16, 2)
  dim3 block(128);              // 4 wave32
  fa_fwd_kernel<<<grid, block, 0, stream>>>(Q, K, V, M, O);
}
